// RecurrentDoubleGNN_26998164423206
// MI455X (gfx1250) — matrix operand order repaired and checked by reading, not compiled, not run on hardware
//
#include <hip/hip_runtime.h>
#include <hip/hip_bf16.h>
#include <math.h>

#define N_NODES 50000
#define N_EDGES 1600000
#define D_IN 14
#define D_HID 64
#define D_OUT 7
#define D_G3  192   // 3*D_HID

typedef __attribute__((ext_vector_type(16))) _Float16 v16h;
typedef __attribute__((ext_vector_type(8)))  float    v8f;

// ---------------- workspace layout (floats) ----------------
#define OFF_DEG  0
#define OFF_H    50048
#define OFF_AGG  (OFF_H   + N_NODES*D_HID)
#define OFF_GI   (OFF_AGG + N_NODES*D_HID)
#define OFF_OUTS OFF_H     // outs reuses h region (h dead after aggregation)

__device__ __forceinline__ float fast_sigmoid(float x) {
    return 1.0f / (1.0f + __expf(-x));          // v_exp_f32 path
}
__device__ __forceinline__ float fast_tanh(float x) {
#if __has_builtin(__builtin_amdgcn_tanhf)
    return __builtin_amdgcn_tanhf(x);           // v_tanh_f32 (CDNA5 TRANS op)
#else
    return tanhf(x);
#endif
}

// ---- init: deg=1 (self loop), agg=0 ----
__global__ __launch_bounds__(256) void init_kernel(float* __restrict__ deg,
                                                   float* __restrict__ agg) {
    int idx = blockIdx.x * 256 + threadIdx.x;
    if (idx < N_NODES * D_HID) agg[idx] = 0.0f;
    if (idx < N_NODES)         deg[idx] = 1.0f;
}

// ---- deg[dst] += 1 over edges ----
__global__ __launch_bounds__(256) void degree_kernel(const int* __restrict__ dst,
                                                     float* __restrict__ deg) {
    int e = blockIdx.x * 256 + threadIdx.x;
    if (e < N_EDGES) atomicAdd(&deg[dst[e]], 1.0f);
}

// ---- dinv = rsqrt(deg) in place (deg >= 1 always) ----
__global__ __launch_bounds__(256) void dinv_kernel(float* __restrict__ deg) {
    int i = blockIdx.x * 256 + threadIdx.x;
    if (i < N_NODES) deg[i] = __frsqrt_rn(deg[i]);
}

// ---- h = x @ W_gcn  (K=14, exact f32; bias added post-aggregation) ----
__global__ __launch_bounds__(256) void gcn_gemm_kernel(const float* __restrict__ x,
                                                       const float* __restrict__ W,
                                                       float* __restrict__ h) {
    int idx = blockIdx.x * 256 + threadIdx.x;       // node*64 + col
    if (idx >= N_NODES * D_HID) return;
    int node = idx >> 6, col = idx & 63;
    const float* xr = x + node * D_IN;
    float acc = 0.0f;
#pragma unroll
    for (int k = 0; k < D_IN; ++k) acc = fmaf(xr[k], W[k * D_HID + col], acc);
    h[idx] = acc;
}

// ---- agg[dst] += h[src] * dinv[src]*dinv[dst]  (edges + self loops) ----
__global__ __launch_bounds__(256) void aggregate_kernel(const float* __restrict__ h,
                                                        const int* __restrict__ src,
                                                        const int* __restrict__ dst,
                                                        const float* __restrict__ dinv,
                                                        float* __restrict__ agg) {
    long long idx = (long long)blockIdx.x * 256 + threadIdx.x;
    const long long total = (long long)(N_EDGES + N_NODES) * 16;
    if (idx >= total) return;
    int e  = (int)(idx >> 4);
    int c4 = (int)(idx & 15) << 2;
    int s, d;
    if (e < N_EDGES) { s = src[e]; d = dst[e]; }
    else             { s = d = e - N_EDGES; }          // self loop
    float nrm = dinv[s] * dinv[d];
    const float4 hv = *(const float4*)(h + (size_t)s * D_HID + c4);
    float* out = agg + (size_t)d * D_HID + c4;
    atomicAdd(out + 0, hv.x * nrm);
    atomicAdd(out + 1, hv.y * nrm);
    atomicAdd(out + 2, hv.z * nrm);
    atomicAdd(out + 3, hv.w * nrm);
}

// ---- GI = (agg + b_gcn) @ W_ih^T + b_ih   via v_wmma_f32_16x16x32_f16 ----
// M=50000 (3125 tiles), N=192 (12 tiles), K=64 (2 x K32). One wave per 16x16 tile.
__global__ __launch_bounds__(256) void gi_gemm_wmma(const float* __restrict__ agg,
                                                    const float* __restrict__ b_gcn,
                                                    const float* __restrict__ W_ih,  // [192,64]
                                                    const float* __restrict__ b_ih,  // [192]
                                                    float* __restrict__ GI) {        // [M,192]
    const int tilesN = D_G3 / 16;                         // 12
    int waveId = blockIdx.x * 8 + (threadIdx.x >> 5);
    int lane   = threadIdx.x & 31;
    int tm = waveId / tilesN;
    int tn = waveId % tilesN;
    if (tm >= N_NODES / 16) return;                       // wave-uniform guard

    int mrow = tm * 16 + (lane & 15);
    int ncol = tn * 16 + (lane & 15);
    int kbA  = (lane >> 4) * 8;                           // A half-wave K base
    int kbB  = (lane >> 4) * 8;                           // B half-wave K base

    v8f c = {};
#pragma unroll
    for (int kc = 0; kc < 2; ++kc) {
        const int Kb = kc * 32;
        v16h a, b;
        // A 16x32 f16 layout: 8 halfs at K[Kb+kbA..+7], 8 halfs at K[Kb+16+kbA..+7]
        // all bases multiple of 4 floats -> aligned float4 (b128) loads
        {
            const float4 a0 = *(const float4*)(agg + (size_t)mrow * D_HID + Kb + kbA);
            const float4 a1 = *(const float4*)(agg + (size_t)mrow * D_HID + Kb + kbA + 4);
            const float4 a2 = *(const float4*)(agg + (size_t)mrow * D_HID + Kb + 16 + kbA);
            const float4 a3 = *(const float4*)(agg + (size_t)mrow * D_HID + Kb + 16 + kbA + 4);
            const float4 g0 = *(const float4*)(b_gcn + Kb + kbA);
            const float4 g1 = *(const float4*)(b_gcn + Kb + kbA + 4);
            const float4 g2 = *(const float4*)(b_gcn + Kb + 16 + kbA);
            const float4 g3 = *(const float4*)(b_gcn + Kb + 16 + kbA + 4);
            a[0]  = (_Float16)(a0.x + g0.x); a[1]  = (_Float16)(a0.y + g0.y);
            a[2]  = (_Float16)(a0.z + g0.z); a[3]  = (_Float16)(a0.w + g0.w);
            a[4]  = (_Float16)(a1.x + g1.x); a[5]  = (_Float16)(a1.y + g1.y);
            a[6]  = (_Float16)(a1.z + g1.z); a[7]  = (_Float16)(a1.w + g1.w);
            a[8]  = (_Float16)(a2.x + g2.x); a[9]  = (_Float16)(a2.y + g2.y);
            a[10] = (_Float16)(a2.z + g2.z); a[11] = (_Float16)(a2.w + g2.w);
            a[12] = (_Float16)(a3.x + g3.x); a[13] = (_Float16)(a3.y + g3.y);
            a[14] = (_Float16)(a3.z + g3.z); a[15] = (_Float16)(a3.w + g3.w);
        }
        // B 32x16 f16 layout: 8 K at Kb+kbB..+7, 8 K at Kb+16+kbB..+7 for column ncol; B[k][n]=W_ih[n][k]
        {
            const float* wrow = W_ih + (size_t)ncol * D_HID + Kb + kbB;
            const float4 b0 = *(const float4*)(wrow + 0);
            const float4 b1 = *(const float4*)(wrow + 4);
            const float4 b2 = *(const float4*)(wrow + 16);
            const float4 b3 = *(const float4*)(wrow + 20);
            b[0]  = (_Float16)b0.x; b[1]  = (_Float16)b0.y;
            b[2]  = (_Float16)b0.z; b[3]  = (_Float16)b0.w;
            b[4]  = (_Float16)b1.x; b[5]  = (_Float16)b1.y;
            b[6]  = (_Float16)b1.z; b[7]  = (_Float16)b1.w;
            b[8]  = (_Float16)b2.x; b[9]  = (_Float16)b2.y;
            b[10] = (_Float16)b2.z; b[11] = (_Float16)b2.w;
            b[12] = (_Float16)b3.x; b[13] = (_Float16)b3.y;
            b[14] = (_Float16)b3.z; b[15] = (_Float16)b3.w;
        }
        c = __builtin_amdgcn_wmma_f32_16x16x32_f16(false, a, false, b,
                                                   (short)0, c, false, false);
    }
    // D layout: VGPR r -> row (tm*16 + r + 8*(lane>=16)), col = ncol
    int rbase = tm * 16 + ((lane >> 4) << 3);
    float bias = b_ih[ncol];
#pragma unroll
    for (int r = 0; r < 8; ++r) {
        GI[(size_t)(rbase + r) * D_G3 + ncol] = c[r] + bias;
    }
}

// ---- sequential GRU scan: one persistent block, W_hh rows in registers ----
// Per-step critical path: issue GI load early (latency hidden under the
// 64-FMA dot), 4 accumulators break the FMA chain, native exp/tanh for gates.
__global__ __launch_bounds__(256) void gru_scan_kernel(const float* __restrict__ GI,   // [N,192] pre-biased
                                                       const float* __restrict__ W_hh, // [192,64]
                                                       const float* __restrict__ b_hh, // [192]
                                                       const float* __restrict__ h0,   // [64]
                                                       float* __restrict__ outs,       // [N,64]
                                                       float* __restrict__ hfinal) {   // [64]
    __shared__ float h_s[D_HID];
    __shared__ float gh_s[D_G3];     // r/z sections hold gi+gh; n section holds gh only
    __shared__ float gin_s[D_HID];   // gi for the n gate (needed separately: r*gh_n)
    const int tid = threadIdx.x;

    float w[D_HID];
    float bh = 0.0f;
    if (tid < D_G3) {
#pragma unroll
        for (int k = 0; k < D_HID; ++k) w[k] = W_hh[tid * D_HID + k];
        bh = b_hh[tid];
    }
    if (tid < D_HID) h_s[tid] = h0[tid];
    __syncthreads();

    for (int t = 0; t < N_NODES; ++t) {
        if (tid < D_G3) {
            if (t + 4 < N_NODES)
                __builtin_prefetch(GI + (size_t)(t + 4) * D_G3 + tid, 0, 1);
            float gi = GI[(size_t)t * D_G3 + tid];        // issued before the dot
            float a0 = 0.f, a1 = 0.f, a2 = 0.f, a3 = 0.f;
#pragma unroll
            for (int k = 0; k < D_HID; k += 4) {
                a0 = fmaf(w[k + 0], h_s[k + 0], a0);
                a1 = fmaf(w[k + 1], h_s[k + 1], a1);
                a2 = fmaf(w[k + 2], h_s[k + 2], a2);
                a3 = fmaf(w[k + 3], h_s[k + 3], a3);
            }
            float gh = bh + ((a0 + a1) + (a2 + a3));
            if (tid < 2 * D_HID) {
                gh_s[tid] = gi + gh;                      // r,z pre-activations fused
            } else {
                gh_s[tid]           = gh;                 // n gate: keep gh separate
                gin_s[tid - 2*D_HID] = gi;
            }
        }
        __syncthreads();
        if (tid < D_HID) {
            float r  = fast_sigmoid(gh_s[tid]);
            float z  = fast_sigmoid(gh_s[D_HID + tid]);
            float n  = fast_tanh(gin_s[tid] + r * gh_s[2*D_HID + tid]);
            float hn = (1.0f - z) * n + z * h_s[tid];
            h_s[tid] = hn;
            outs[(size_t)t * D_HID + tid] = hn;
        }
        __syncthreads();
    }
    if (tid < D_HID) hfinal[tid] = h_s[tid];
}

// ---- y = outs @ W_fc + b_fc ----
__global__ __launch_bounds__(256) void fc_kernel(const float* __restrict__ outs,
                                                 const float* __restrict__ W_fc, // [64,7]
                                                 const float* __restrict__ b_fc, // [7]
                                                 float* __restrict__ y) {        // [N,7]
    int idx = blockIdx.x * 256 + threadIdx.x;
    if (idx >= N_NODES * D_OUT) return;
    int node = idx / D_OUT, o = idx % D_OUT;
    const float* hr = outs + (size_t)node * D_HID;
    float acc = b_fc[o];
#pragma unroll
    for (int c = 0; c < D_HID; ++c) acc = fmaf(hr[c], W_fc[c * D_OUT + o], acc);
    y[idx] = acc;
}

extern "C" void kernel_launch(void* const* d_in, const int* in_sizes, int n_in,
                              void* d_out, int out_size, void* d_ws, size_t ws_size,
                              hipStream_t stream) {
    const float* x     = (const float*)d_in[0];
    const int*   eidx  = (const int*)  d_in[1];
    const float* h0    = (const float*)d_in[2];
    const float* W_gcn = (const float*)d_in[3];
    const float* b_gcn = (const float*)d_in[4];
    const float* W_ih  = (const float*)d_in[5];
    const float* W_hh  = (const float*)d_in[6];
    const float* b_ih  = (const float*)d_in[7];
    const float* b_hh  = (const float*)d_in[8];
    const float* W_fc  = (const float*)d_in[9];
    const float* b_fc  = (const float*)d_in[10];

    const int* src = eidx;
    const int* dst = eidx + N_EDGES;

    float* ws   = (float*)d_ws;
    float* deg  = ws + OFF_DEG;
    float* h    = ws + OFF_H;
    float* agg  = ws + OFF_AGG;
    float* GI   = ws + OFF_GI;
    float* outs = ws + OFF_OUTS;

    float* y      = (float*)d_out;                 // [N,7]
    float* hfinal = (float*)d_out + N_NODES * D_OUT;

    init_kernel<<<(N_NODES * D_HID + 255) / 256, 256, 0, stream>>>(deg, agg);
    degree_kernel<<<(N_EDGES + 255) / 256, 256, 0, stream>>>(dst, deg);
    dinv_kernel<<<(N_NODES + 255) / 256, 256, 0, stream>>>(deg);
    gcn_gemm_kernel<<<(N_NODES * D_HID + 255) / 256, 256, 0, stream>>>(x, W_gcn, h);
    {
        long long total = (long long)(N_EDGES + N_NODES) * 16;
        aggregate_kernel<<<(int)((total + 255) / 256), 256, 0, stream>>>(h, src, dst, deg, agg);
    }
    {
        int tiles = (N_NODES / 16) * (D_G3 / 16);          // 37500 waves
        gi_gemm_wmma<<<(tiles + 7) / 8, 256, 0, stream>>>(agg, b_gcn, W_ih, b_ih, GI);
    }
    gru_scan_kernel<<<1, 256, 0, stream>>>(GI, W_hh, b_hh, h0, outs, hfinal);
    fc_kernel<<<(N_NODES * D_OUT + 255) / 256, 256, 0, stream>>>(outs, W_fc, b_fc, y);
}
